// SwinTransformerBlock_12644383719485
// MI455X (gfx1250) — matrix operand order repaired and checked by reading, not compiled, not run on hardware
//
#include <hip/hip_runtime.h>
#include <hip/hip_bf16.h>
#include <math.h>

typedef __attribute__((ext_vector_type(16))) _Float16 v16h;
typedef __attribute__((ext_vector_type(8)))  float    v8f;

namespace {
constexpr int HIMG = 56, WS = 7, SHIFT = 3, HEADS = 4, DIM = 128, HD = 32;
constexpr int NTOK = 49;        // tokens per window
constexpr int NPAD = 64;        // padded to 4x16 WMMA tiles
constexpr int NWIN = 64;        // windows per image
constexpr int HID  = 512;
constexpr int TOKS = HIMG * HIMG;
constexpr float QSCALE = 0.17677669529663687f;  // 32^-0.5

// f16 weight cache layout inside d_ws (element offsets)
constexpr int OFF_QKV  = 0;            // 384x128
constexpr int OFF_PROJ = 49152;        // 128x128
constexpr int OFF_FC1  = 65536;        // 512x128
constexpr int OFF_FC2  = 131072;       // 128x512
constexpr int W_TOTAL  = 196608;

// Kernel A LDS layout (byte offsets); total 150016 B -> 2 workgroups/WGP
constexpr int A_STATS = 0;             // 2560 B
constexpr int A_XLN   = 2560;          // 64x128 f16 = 16384
constexpr int A_QK    = 18944;         // 64x256 f16 (Q cols 0-127, K cols 128-255)
constexpr int A_VT    = 51712;         // 128x64 f16 (V transposed) = 16384
constexpr int A_SCO   = 68096;         // 4x64x64 f32 = 65536 (aliased: xraw, probs)
constexpr int A_OSM   = 133632;        // 64x128 f16 = 16384
constexpr int A_TOTAL = 150016;

// Kernel B LDS layout; total 117248 B -> 2 workgroups/WGP
constexpr int B_TOTAL = 117248;
}

// ---------------- WMMA helpers (CDNA5 16x16x32 f16 -> f32) ----------------

static __device__ inline v8f wmma16(v16h a, v16h b, v8f c) {
  // (neg_a, A, neg_b, B, c_mod, C, reuse_a, reuse_b)
  return __builtin_amdgcn_wmma_f32_16x16x32_f16(false, a, false, b, (short)0, c,
                                                false, false);
}

// A fragment, 16x32 f16 tile, row-major source (LDS, f16), in contiguous k order:
// lanes 0-15: M=lane, K {0..15}; lanes 16-31: M=lane-16, K {16..31}
// -> one contiguous 32B run per lane (matches the B fragment loaders).
static __device__ inline v16h ld_a_lds(const _Float16* base, int lda, int kbase,
                                       int lane) {
  const int m = lane & 15;
  const int sel = (lane >> 4) & 1;
  const _Float16* row = base + m * lda + kbase;
  v16h a;
#pragma unroll
  for (int g = 0; g < 8; ++g) {
    int k = sel * 16 + (g << 1);
    a[2 * g]     = row[k];
    a[2 * g + 1] = row[k + 1];
  }
  return a;
}

// B fragment where logical B[k][n] = W[(n0+n)*ldw + k0+k], W row-major f16.
// 16 consecutive halves per lane (merges into contiguous 32B loads).
static __device__ inline v16h ld_bT_h(const _Float16* __restrict__ W, int ldw,
                                      int n0, int k0, int lane) {
  const int n = lane & 15;
  const int ko = ((lane >> 4) & 1) * 16;
  const _Float16* p = W + (size_t)(n0 + n) * ldw + k0 + ko;
  v16h b;
#pragma unroll
  for (int i = 0; i < 16; ++i) b[i] = p[i];
  return b;
}

// Same contiguous pattern from f16 LDS (K^T and V^T fragments).
static __device__ inline v16h ld_bT_lds(const _Float16* src, int ld, int n0,
                                        int k0, int lane) {
  const int n = lane & 15;
  const int ko = ((lane >> 4) & 1) * 16;
  const _Float16* p = src + (n0 + n) * ld + k0 + ko;
  v16h b;
#pragma unroll
  for (int i = 0; i < 16; ++i) b[i] = p[i];
  return b;
}

// roll(-3,-3) + window-partition source token (also the residual destination)
static __device__ inline int src_token(int b, int wr, int wc, int t) {
  const int i = t / WS, j = t % WS;
  const int hh = (wr * WS + i + SHIFT) % HIMG;
  const int ww = (wc * WS + j + SHIFT) % HIMG;
  return b * TOKS + hh * HIMG + ww;
}

// shift-mask region id along one axis of the (un-rolled) mask image
static __device__ inline int regid(int p) { return p < 49 ? 0 : (p < 53 ? 1 : 2); }

// =====================================================================
// Kernel 0: one-shot f32 -> f16 conversion of all weight matrices into d_ws.
// =====================================================================
__global__ void cvt_weights_kernel(const float* __restrict__ qkv_w,
                                   const float* __restrict__ proj_w,
                                   const float* __restrict__ fc1_w,
                                   const float* __restrict__ fc2_w,
                                   _Float16* __restrict__ wh) {
  const int i = blockIdx.x * 256 + threadIdx.x;
  if (i >= W_TOTAL) return;
  float v;
  if (i < OFF_PROJ)      v = qkv_w[i - OFF_QKV];
  else if (i < OFF_FC1)  v = proj_w[i - OFF_PROJ];
  else if (i < OFF_FC2)  v = fc1_w[i - OFF_FC1];
  else                   v = fc2_w[i - OFF_FC2];
  wh[i] = (_Float16)v;
}

// =====================================================================
// Kernel A: one workgroup (8 wave32) per 7x7 window.
// LN1 -> QKV -> Q K^T (+rel-bias +shift-mask) -> softmax (in-place f16
// overlay) -> P V (V^T in LDS) -> proj -> un-roll scatter + residual.
// 150 KB LDS => two workgroups per WGP (4 waves/SIMD) for latency hiding.
// =====================================================================
__global__ void swin_attn_kernel(const float* __restrict__ x,
                                 const float* __restrict__ g1,
                                 const float* __restrict__ b1,
                                 const _Float16* __restrict__ wh,
                                 const float* __restrict__ qkv_b,
                                 const float* __restrict__ rpb,
                                 const float* __restrict__ proj_b,
                                 float* __restrict__ out) {
  extern __shared__ char smem[];
  float*    mean_s = (float*)(smem + A_STATS);       //  64 f32
  float*    rstd_s = mean_s + NPAD;                  //  64 f32
  float*    psum   = rstd_s + NPAD;                  // 4x64 f32
  float*    psum2  = psum + 256;                     // 4x64 f32
  _Float16* xln    = (_Float16*)(smem + A_XLN);      //  64x128 f16
  _Float16* qks    = (_Float16*)(smem + A_QK);       //  64x256 f16 (Q|K)
  _Float16* vT     = (_Float16*)(smem + A_VT);       // 128x64 f16 (V^T)
  float*    sco    = (float*)(smem + A_SCO);         // 4x64x64 f32 scores
  float*    xraw   = (float*)(smem + A_SCO);         // alias: 64x128 f32 (ph 0-1)
  _Float16* prbS   = (_Float16*)(smem + A_SCO);      // alias: probs overlay rows
  _Float16* osm    = (_Float16*)(smem + A_OSM);      //  64x128 f16

  const int tid = threadIdx.x;
  const int wave = tid >> 5, lane = tid & 31;
  const int win = blockIdx.x;
  const int b = win >> 6, widx = win & 63;
  const int wr = widx >> 3, wc = widx & 7;

  const _Float16* qkv_wh  = wh + OFF_QKV;
  const _Float16* proj_wh = wh + OFF_PROJ;
  __builtin_prefetch(qkv_wh + tid * 384, 0, 0);   // global_prefetch_b8

  // ---- phase 0: gather rolled window tile into LDS (aliased with scores) ----
  for (int idx = tid; idx < NPAD * DIM; idx += 256) {
    const int t = idx >> 7, c = idx & 127;
    float v = 0.f;
    if (t < NTOK) v = x[(size_t)src_token(b, wr, wc, t) * DIM + c];
    xraw[idx] = v;
  }
  __syncthreads();

  // ---- phase 1: LayerNorm1 stats, 4-way parallel per token ----
  {
    const int t = tid & 63, p = tid >> 6;
    const float* row = xraw + t * DIM + p * 32;
    float s = 0.f, s2 = 0.f;
#pragma unroll
    for (int c = 0; c < 32; ++c) { float v = row[c]; s += v; s2 += v * v; }
    psum[p * 64 + t] = s;
    psum2[p * 64 + t] = s2;
  }
  __syncthreads();
  if (tid < NPAD) {
    float s  = psum[tid] + psum[64 + tid] + psum[128 + tid] + psum[192 + tid];
    float s2 = psum2[tid] + psum2[64 + tid] + psum2[128 + tid] + psum2[192 + tid];
    float m = s * (1.0f / DIM);
    float var = s2 * (1.0f / DIM) - m * m;
    mean_s[tid] = m;
    rstd_s[tid] = rsqrtf(var + 1e-5f);
  }
  __syncthreads();
  for (int idx = tid; idx < NPAD * DIM; idx += 256) {
    const int t = idx >> 7, c = idx & 127;
    float v = (t < NTOK)
        ? (xraw[idx] - mean_s[t]) * rstd_s[t] * g1[c] + b1[c]
        : 0.f;
    xln[idx] = (_Float16)v;
  }
  __syncthreads();

  // ---- phase 2: QKV = xln * qkv_w^T + qkv_b (Q scaled; V stored transposed) ----
  for (int it = 0; it < 12; ++it) {
    const int tile = it * 8 + wave;        // 96 tiles: 4 M x 24 N
    const int mt = tile / 24, nt = tile % 24;
    const int n0 = nt * 16;
    const float bias = qkv_b[n0 + (lane & 15)];
    v8f acc = {bias, bias, bias, bias, bias, bias, bias, bias};
#pragma unroll
    for (int kk = 0; kk < 4; ++kk) {
      v16h a  = ld_a_lds(xln + mt * 16 * DIM, DIM, kk * 32, lane);
      v16h bm = ld_bT_h(qkv_wh, DIM, n0, kk * 32, lane);
      acc = wmma16(a, bm, acc);
    }
    const int n = n0 + (lane & 15);
    if (n0 < 2 * DIM) {                    // Q (scaled) and K rows
      const float sc = (n0 < DIM) ? QSCALE : 1.0f;
#pragma unroll
      for (int r = 0; r < 8; ++r) {
        const int m = mt * 16 + r + ((lane >> 4) << 3);
        qks[m * 256 + n] = (_Float16)(acc[r] * sc);
      }
    } else {                               // V -> transposed [chan][token]
      const int ch = n - 2 * DIM;
#pragma unroll
      for (int r = 0; r < 8; ++r) {
        const int m = mt * 16 + r + ((lane >> 4) << 3);
        vT[ch * 64 + m] = (_Float16)acc[r];
      }
    }
  }
  __syncthreads();

  // ---- phase 3: S = Q K^T + rel_bias + shift_mask ----
  for (int it = 0; it < 8; ++it) {
    const int tile = it * 8 + wave;        // 64 tiles: 4 heads x 4 M x 4 N
    const int head = tile >> 4;
    const int mt = (tile >> 2) & 3, nt = tile & 3;
    v8f acc = {};
    {
      v16h a  = ld_a_lds(qks + (mt * 16) * 256 + head * HD, 256, 0, lane);
      v16h bm = ld_bT_lds(qks + DIM + head * HD, 256, nt * 16, 0, lane);
      acc = wmma16(a, bm, acc);
    }
    const int n = nt * 16 + (lane & 15);
#pragma unroll
    for (int r = 0; r < 8; ++r) {
      const int m = mt * 16 + r + ((lane >> 4) << 3);
      float val;
      if (n >= NTOK) {
        val = -1e30f;                       // padded key column
      } else if (m >= NTOK) {
        val = 0.f;                          // padded query row (unused)
      } else {
        const int im = m / WS, jm = m % WS, in_ = n / WS, jn = n % WS;
        const int ridx = (im - in_ + 6) * 13 + (jm - jn + 6);
        const float bias = rpb[ridx * HEADS + head];
        const int rm = regid(wr * WS + im) * 3 + regid(wc * WS + jm);
        const int rn = regid(wr * WS + in_) * 3 + regid(wc * WS + jn);
        val = acc[r] + bias + ((rm != rn) ? -100.f : 0.f);
      }
      sco[(head * 64 + m) * 64 + n] = val;
    }
  }
  __syncthreads();

  // ---- phase 4: softmax, one row/thread; f16 probs overlaid on own f32 row ----
  {
    const int head = tid >> 6, m = tid & 63;
    float* row = sco + (head * 64 + m) * 64;
    _Float16* prow = (_Float16*)row;       // in-place overlay (thread-owned row)
    if (m < NTOK) {
      float mx = -1e30f;
      for (int n = 0; n < NTOK; ++n) mx = fmaxf(mx, row[n]);
      float s = 0.f;
      for (int n = 0; n < NTOK; ++n) {     // read f32 before overwriting bytes
        float e = expf(row[n] - mx);
        s += e;
        prow[n] = (_Float16)e;
      }
      const float inv = 1.f / s;
      for (int n = 0; n < NTOK; ++n) prow[n] = (_Float16)((float)prow[n] * inv);
      for (int n = NTOK; n < 64; ++n) prow[n] = (_Float16)0.f;
    } else {
      for (int n = 0; n < 64; ++n) prow[n] = (_Float16)0.f;
    }
  }
  __syncthreads();

  // ---- phase 5: O = P V (per head) via V^T, concat heads into 64x128 ----
  for (int it = 0; it < 4; ++it) {
    const int tile = it * 8 + wave;        // 32 tiles: 4 heads x 4 M x 2 N
    const int head = tile >> 3;
    const int mt = (tile >> 1) & 3, nt = tile & 1;
    v8f acc = {};
#pragma unroll
    for (int kk = 0; kk < 2; ++kk) {
      // probs rows live in the first 64 halves of each 256B score row
      v16h a  = ld_a_lds(prbS + (size_t)(head * 64 + mt * 16) * 128, 128,
                         kk * 32, lane);
      v16h bm = ld_bT_lds(vT + head * HD * 64, 64, nt * 16, kk * 32, lane);
      acc = wmma16(a, bm, acc);
    }
    const int col = head * HD + nt * 16 + (lane & 15);
#pragma unroll
    for (int r = 0; r < 8; ++r) {
      const int m = mt * 16 + r + ((lane >> 4) << 3);
      osm[m * DIM + col] = (_Float16)acc[r];
    }
  }
  __syncthreads();

  // ---- phase 6: proj + un-roll scatter + residual add into d_out ----
  for (int it = 0; it < 4; ++it) {
    const int tile = it * 8 + wave;        // 32 tiles: 4 M x 8 N
    const int mt = tile >> 3, nt = tile & 7;
    const float bias = proj_b[nt * 16 + (lane & 15)];
    v8f acc = {bias, bias, bias, bias, bias, bias, bias, bias};
#pragma unroll
    for (int kk = 0; kk < 4; ++kk) {
      v16h a  = ld_a_lds(osm + mt * 16 * DIM, DIM, kk * 32, lane);
      v16h bm = ld_bT_h(proj_wh, DIM, nt * 16, kk * 32, lane);
      acc = wmma16(a, bm, acc);
    }
    const int col = nt * 16 + (lane & 15);
#pragma unroll
    for (int r = 0; r < 8; ++r) {
      const int m = mt * 16 + r + ((lane >> 4) << 3);
      if (m < NTOK) {
        const size_t o = (size_t)src_token(b, wr, wc, m) * DIM + col;
        out[o] = x[o] + acc[r];            // x re-read is L2-resident
      }
    }
  }
}

// =====================================================================
// Kernel B: one workgroup per 64-token tile.
// LN2 -> fc1 (128->512) -> exact GELU -> fc2 (512->128) -> residual.
// =====================================================================
__global__ void swin_mlp_kernel(const float* __restrict__ g2,
                                const float* __restrict__ b2,
                                const _Float16* __restrict__ wh,
                                const float* __restrict__ fc1_b,
                                const float* __restrict__ fc2_b,
                                float* __restrict__ out) {
  extern __shared__ char smem[];
  float*    mean_s = (float*)smem;                    //  64 f32
  float*    rstd_s = mean_s + 64;                     //  64 f32
  float*    psum   = rstd_s + 64;                     // 4x64 f32
  float*    psum2  = psum + 256;                      // 4x64 f32
  float*    xraw   = (float*)(smem + 2560);           //  64x128 f32 (x')
  _Float16* xln    = (_Float16*)(smem + 35328);       //  64x128 f16
  _Float16* hid    = (_Float16*)(smem + 51712);       //  64x512 f16

  const int tid = threadIdx.x;
  const int wave = tid >> 5, lane = tid & 31;
  const size_t tok0 = (size_t)blockIdx.x * 64;

  const _Float16* fc1_wh = wh + OFF_FC1;
  const _Float16* fc2_wh = wh + OFF_FC2;
  __builtin_prefetch(fc1_wh + tid * 256, 0, 0);       // global_prefetch_b8

  // ---- stage x' tile ----
  for (int idx = tid; idx < 64 * DIM; idx += 256)
    xraw[idx] = out[tok0 * DIM + idx];
  __syncthreads();

  // ---- LN2 stats, 4-way parallel per token ----
  {
    const int t = tid & 63, p = tid >> 6;
    const float* row = xraw + t * DIM + p * 32;
    float s = 0.f, s2 = 0.f;
#pragma unroll
    for (int c = 0; c < 32; ++c) { float v = row[c]; s += v; s2 += v * v; }
    psum[p * 64 + t] = s;
    psum2[p * 64 + t] = s2;
  }
  __syncthreads();
  if (tid < 64) {
    float s  = psum[tid] + psum[64 + tid] + psum[128 + tid] + psum[192 + tid];
    float s2 = psum2[tid] + psum2[64 + tid] + psum2[128 + tid] + psum2[192 + tid];
    float m = s * (1.0f / DIM);
    float var = s2 * (1.0f / DIM) - m * m;
    mean_s[tid] = m;
    rstd_s[tid] = rsqrtf(var + 1e-5f);
  }
  __syncthreads();
  for (int idx = tid; idx < 64 * DIM; idx += 256) {
    const int t = idx >> 7, c = idx & 127;
    xln[idx] = (_Float16)((xraw[idx] - mean_s[t]) * rstd_s[t] * g2[c] + b2[c]);
  }
  __syncthreads();

  // ---- fc1 + exact GELU : 128 tiles (4 M x 32 N) ----
  for (int it = 0; it < 16; ++it) {
    const int tile = it * 8 + wave;
    const int mt = tile >> 5, nt = tile & 31;
    const float bias = fc1_b[nt * 16 + (lane & 15)];
    v8f acc = {bias, bias, bias, bias, bias, bias, bias, bias};
#pragma unroll
    for (int kk = 0; kk < 4; ++kk) {
      v16h a  = ld_a_lds(xln + mt * 16 * DIM, DIM, kk * 32, lane);
      v16h bm = ld_bT_h(fc1_wh, DIM, nt * 16, kk * 32, lane);
      acc = wmma16(a, bm, acc);
    }
    const int col = nt * 16 + (lane & 15);
#pragma unroll
    for (int r = 0; r < 8; ++r) {
      const int m = mt * 16 + r + ((lane >> 4) << 3);
      float v = acc[r];
      v = 0.5f * v * (1.f + erff(v * 0.70710678118654752f));  // exact GELU
      hid[m * HID + col] = (_Float16)v;
    }
  }
  __syncthreads();

  // ---- fc2 + residual : 32 tiles (4 M x 8 N), K = 512 ----
  for (int it = 0; it < 4; ++it) {
    const int tile = it * 8 + wave;
    const int mt = tile >> 3, nt = tile & 7;
    const float bias = fc2_b[nt * 16 + (lane & 15)];
    v8f acc = {bias, bias, bias, bias, bias, bias, bias, bias};
#pragma unroll
    for (int kk = 0; kk < 16; ++kk) {
      v16h a  = ld_a_lds(hid + mt * 16 * HID, HID, kk * 32, lane);
      v16h bm = ld_bT_h(fc2_wh, HID, nt * 16, kk * 32, lane);
      acc = wmma16(a, bm, acc);
    }
    const int col = nt * 16 + (lane & 15);
#pragma unroll
    for (int r = 0; r < 8; ++r) {
      const int m = mt * 16 + r + ((lane >> 4) << 3);
      const size_t o = (tok0 + m) * DIM + col;
      out[o] = xraw[m * DIM + col] + acc[r];     // final = x' + mlp(x')
    }
  }
}

// =====================================================================
extern "C" void kernel_launch(void* const* d_in, const int* in_sizes, int n_in,
                              void* d_out, int out_size, void* d_ws,
                              size_t ws_size, hipStream_t stream) {
  (void)n_in; (void)out_size; (void)ws_size;
  const float* x      = (const float*)d_in[0];
  const float* g1     = (const float*)d_in[1];
  const float* b1     = (const float*)d_in[2];
  const float* qkv_w  = (const float*)d_in[3];
  const float* qkv_b  = (const float*)d_in[4];
  const float* rpb    = (const float*)d_in[5];
  const float* proj_w = (const float*)d_in[6];
  const float* proj_b = (const float*)d_in[7];
  const float* g2     = (const float*)d_in[8];
  const float* b2     = (const float*)d_in[9];
  const float* fc1_w  = (const float*)d_in[10];
  const float* fc1_b  = (const float*)d_in[11];
  const float* fc2_w  = (const float*)d_in[12];
  const float* fc2_b  = (const float*)d_in[13];
  float* out = (float*)d_out;
  _Float16* wh = (_Float16*)d_ws;

  const int Bimg = in_sizes[0] / (TOKS * DIM);
  const int nwin = Bimg * NWIN;
  const int total_tokens = Bimg * TOKS;

  // CDNA5 WGP has 320KB LDS; raise the per-kernel dynamic-LDS cap.
  hipFuncSetAttribute((const void*)swin_attn_kernel,
                      hipFuncAttributeMaxDynamicSharedMemorySize, A_TOTAL);
  hipFuncSetAttribute((const void*)swin_mlp_kernel,
                      hipFuncAttributeMaxDynamicSharedMemorySize, B_TOTAL);

  cvt_weights_kernel<<<(W_TOTAL + 255) / 256, 256, 0, stream>>>(
      qkv_w, proj_w, fc1_w, fc2_w, wh);
  swin_attn_kernel<<<nwin, 256, (size_t)A_TOTAL, stream>>>(
      x, g1, b1, wh, qkv_b, rpb, proj_b, out);
  swin_mlp_kernel<<<total_tokens / 64, 256, (size_t)B_TOTAL, stream>>>(
      g2, b2, wh, fc1_b, fc2_b, out);
}
